// MultiQueryAttention_50972671869427
// MI455X (gfx1250) — compile-verified, matrix operand order checked by reading, not run on hardware
//
#include <hip/hip_runtime.h>
#include <hip/hip_bf16.h>

// ---------------------------------------------------------------------------
// MQA forward for MI455X (gfx1250, wave32, WMMA).
// Outputs: out [2,2048,1024] fp32, attn [2,16,2048,2048] fp32 (concatenated).
// Strategy: bf16 WMMA (f32 accum) everywhere; attn write is the HBM roofline
// term (536 MB), streamed with non-temporal stores.
// ---------------------------------------------------------------------------

typedef __bf16 bf16_t;
typedef __attribute__((ext_vector_type(16))) __bf16 v16bf;
typedef __attribute__((ext_vector_type(8)))  __bf16 v8bf;
typedef __attribute__((ext_vector_type(8)))  float  v8f;

#define S_LEN 2048
#define BTCH  2
#define HEADS 16
#define DK    64
#define DMODEL 1024
#define MROWS (BTCH * S_LEN)   // 4096

__device__ __forceinline__ v8f zero8() {
    v8f z;
#pragma unroll
    for (int i = 0; i < 8; ++i) z[i] = 0.0f;
    return z;
}

__device__ __forceinline__ v16bf cat8(v8bf lo, v8bf hi) {
    v16bf r;
#pragma unroll
    for (int i = 0; i < 8; ++i) { r[i] = lo[i]; r[i + 8] = hi[i]; }
    return r;
}

// A-fragment from fp32 row (row-major, reads cols [0,8) and [16,24) rel. ptr)
__device__ __forceinline__ v16bf loadA_f32(const float* p) {
    v16bf r;
#pragma unroll
    for (int i = 0; i < 8; ++i) r[i] = (bf16_t)p[i];
#pragma unroll
    for (int i = 0; i < 8; ++i) r[i + 8] = (bf16_t)p[16 + i];
    return r;
}

__device__ __forceinline__ v8f wmma_bf16(v16bf a, v16bf b, v8f c) {
    return __builtin_amdgcn_wmma_f32_16x16x32_bf16(
        false, a, false, b, (short)0, c, false, false);
}

// ---------------------------------------------------------------------------
// Generic GEMM: C[M,N] = A[M,K] @ W[K,N] + bias.  BK = 32.
// A may be fp32 or bf16; C may be bf16 (scratch) or fp32 (final out).
// 256 threads = 8 waves; wave (wr,wc) owns an MT*16 x NT*16 sub-tile.
// All dims assumed multiples of the tile sizes (true for this problem).
// ---------------------------------------------------------------------------
template <int BM, int BN, int WGM, int WGN, int MT, int NT, bool A_BF16, bool OUT_BF16>
__global__ __launch_bounds__(256) void gemm_bias_wmma(
    const void* __restrict__ Av, const float* __restrict__ W,
    const float* __restrict__ bias, void* __restrict__ Cv,
    int M, int N, int K)
{
    static_assert(WGM * WGN == 8, "8 waves");
    static_assert(BM == WGM * MT * 16 && BN == WGN * NT * 16, "tiling");
    constexpr int BK = 32;

    __shared__ bf16_t As[BM * BK];      // [BM][32] row-major
    __shared__ bf16_t Bs[BN * BK];      // W tile transposed: [n][k]

    const int tid  = threadIdx.x;
    const int lane = tid & 31;
    const int wave = tid >> 5;
    const int wr = wave / WGN, wc = wave % WGN;
    const int m0 = blockIdx.y * BM;
    const int n0 = blockIdx.x * BN;
    const int mrow = lane & 15;
    const int kb   = (lane >> 4) * 8;

    const float*  Af = (const float*)Av;
    const bf16_t* Ab = (const bf16_t*)Av;

    v8f acc[MT][NT];
#pragma unroll
    for (int mt = 0; mt < MT; ++mt)
#pragma unroll
        for (int nt = 0; nt < NT; ++nt) acc[mt][nt] = zero8();

    for (int k0 = 0; k0 < K; k0 += BK) {
        // stage A tile (convert to bf16), coalesced
        constexpr int AIT = BM * BK / 256;
#pragma unroll
        for (int i = 0; i < AIT; ++i) {
            int idx = tid + i * 256;
            int r = idx >> 5, c = idx & 31;
            float v = A_BF16 ? (float)Ab[(size_t)(m0 + r) * K + k0 + c]
                             : Af[(size_t)(m0 + r) * K + k0 + c];
            As[r * BK + c] = (bf16_t)v;
        }
        // stage W tile transposed (B-fragment layout mirrors A)
        constexpr int BIT = BN * BK / 256;
#pragma unroll
        for (int i = 0; i < BIT; ++i) {
            int idx = tid + i * 256;
            int n = idx % BN, kk = idx / BN;
            Bs[n * BK + kk] = (bf16_t)W[(size_t)(k0 + kk) * N + n0 + n];
        }
        __syncthreads();

        v16bf af[MT], bfr[NT];
#pragma unroll
        for (int mt = 0; mt < MT; ++mt) {
            const bf16_t* p = &As[((wr * MT + mt) * 16 + mrow) * BK + kb];
            af[mt] = cat8(*(const v8bf*)p, *(const v8bf*)(p + 16));
        }
#pragma unroll
        for (int nt = 0; nt < NT; ++nt) {
            const bf16_t* p = &Bs[((wc * NT + nt) * 16 + mrow) * BK + kb];
            bfr[nt] = cat8(*(const v8bf*)p, *(const v8bf*)(p + 16));
        }
#pragma unroll
        for (int mt = 0; mt < MT; ++mt)
#pragma unroll
            for (int nt = 0; nt < NT; ++nt)
                acc[mt][nt] = wmma_bf16(af[mt], bfr[nt], acc[mt][nt]);
        __syncthreads();
    }

    // epilogue: C layout lane=(col), VGPR r -> row r + 8*(lane>>4)
    const int rhi = (lane >> 4) * 8;
#pragma unroll
    for (int mt = 0; mt < MT; ++mt)
#pragma unroll
        for (int nt = 0; nt < NT; ++nt) {
            int gn = n0 + (wc * NT + nt) * 16 + mrow;
            float bv = bias[gn];
#pragma unroll
            for (int r = 0; r < 8; ++r) {
                int gm = m0 + (wr * MT + mt) * 16 + rhi + r;
                float val = acc[mt][nt][r] + bv;
                if (OUT_BF16) ((bf16_t*)Cv)[(size_t)gm * N + gn] = (bf16_t)val;
                else          ((float*)Cv)[(size_t)gm * N + gn] = val;
            }
        }
}

// V [B*S][64] bf16  ->  Vt [B][64][S] bf16 (so P@V B-fragments are contiguous)
__global__ __launch_bounds__(256) void transpose_v_kernel(
    const bf16_t* __restrict__ v, bf16_t* __restrict__ vt)
{
    int idx = blockIdx.x * 256 + threadIdx.x;      // 0 .. 262143
    int b   = idx >> 17;                           // S_LEN*DK = 131072
    int rem = idx & (S_LEN * DK - 1);
    int s = rem >> 6, d = rem & 63;
    vt[((size_t)b * DK + d) * S_LEN + s] = v[idx];
}

// ---------------------------------------------------------------------------
// Attention: one block = (batch b, head h, 32-query tile).
// scores tile [32][2048] fp32 lives in dynamic LDS (256KB of 320KB/WGP).
//   1) S = (Q @ K^T) * 1/8 via WMMA  (K rows contiguous -> direct global B-frags)
//   2) exact softmax rows (max, exp, 1/sum)
//   3) coalesced NT-stream of attn tile to global (dominant HBM term)
//   4) heads = P @ V via WMMA (A-frags converted fp32->bf16 from LDS)
// ---------------------------------------------------------------------------
__global__ __launch_bounds__(256) void mqa_attn_kernel(
    const bf16_t* __restrict__ q_bf,    // [B*S][DMODEL]
    const bf16_t* __restrict__ k_bf,    // [B*S][DK]
    const bf16_t* __restrict__ vt_bf,   // [B][DK][S]
    float* __restrict__ attn_out,       // [B][H][S][S]
    bf16_t* __restrict__ heads_bf)      // [B*S][DMODEL]  (B,S,H,dk fused)
{
    extern __shared__ char dynsm[];
    float*  sc   = (float*)dynsm;                                  // [32][2048]
    bf16_t* qa   = (bf16_t*)(dynsm + 32 * S_LEN * 4);              // [32][64]
    float*  red  = (float*)(dynsm + 32 * S_LEN * 4 + 32 * DK * 2); // [32][8]
    float*  rmax = red + 256;                                      // [32]
    float*  rinv = rmax + 32;                                      // [32]

    const int tid = threadIdx.x, lane = tid & 31, wave = tid >> 5;
    const int h = blockIdx.y, b = blockIdx.z;
    const int q0 = blockIdx.x * 32;
    const int mrow = lane & 15;
    const int kb   = (lane >> 4) * 8;

    // ---- load Q tile (bf16, already projected) into LDS ----
#pragma unroll
    for (int i = 0; i < 8; ++i) {
        int idx = tid + i * 256;                 // 32*64 = 2048 elems
        int r = idx >> 6, c = idx & 63;
        qa[idx] = q_bf[(size_t)(b * S_LEN + q0 + r) * DMODEL + h * DK + c];
    }
    __syncthreads();

    // ---- preload Q A-fragments: 2 m-tiles x 2 k-steps ----
    v16bf aQ[2][2];
#pragma unroll
    for (int mt = 0; mt < 2; ++mt)
#pragma unroll
        for (int ks = 0; ks < 2; ++ks) {
            const bf16_t* p = &qa[(mt * 16 + mrow) * DK + ks * 32 + kb];
            aQ[mt][ks] = cat8(*(const v8bf*)p, *(const v8bf*)(p + 16));
        }

    // ---- scores: each wave owns 16 consecutive 16-key tiles ----
    for (int i = 0; i < 16; ++i) {
        int ntile = wave * 16 + i;
        int key   = ntile * 16 + mrow;           // lane = column of C = key idx
        const bf16_t* kp = k_bf + (size_t)(b * S_LEN + key) * DK;
        v16bf bf0 = cat8(*(const v8bf*)(kp + kb),      *(const v8bf*)(kp + kb + 16));
        v16bf bf1 = cat8(*(const v8bf*)(kp + 32 + kb), *(const v8bf*)(kp + 48 + kb));
#pragma unroll
        for (int mt = 0; mt < 2; ++mt) {
            v8f c = zero8();
            c = wmma_bf16(aQ[mt][0], bf0, c);
            c = wmma_bf16(aQ[mt][1], bf1, c);
            int rbase = mt * 16 + (lane >> 4) * 8;
            int col   = ntile * 16 + mrow;
#pragma unroll
            for (int r = 0; r < 8; ++r)
                sc[(rbase + r) * S_LEN + col] = c[r] * 0.125f;  // 1/sqrt(64)
        }
    }
    __syncthreads();

    // ---- softmax: 8 threads per row, interleaved columns ----
    const int row = tid >> 3;
    const int seg = tid & 7;
    float mx = -3.4e38f;
    for (int j = 0; j < 256; ++j)
        mx = fmaxf(mx, sc[row * S_LEN + seg + 8 * j]);
    red[row * 8 + seg] = mx;
    __syncthreads();
    if (seg == 0) {
        float m = red[row * 8];
        for (int j = 1; j < 8; ++j) m = fmaxf(m, red[row * 8 + j]);
        rmax[row] = m;
    }
    __syncthreads();
    float m = rmax[row];
    float sum = 0.0f;
    for (int j = 0; j < 256; ++j) {
        int idx = row * S_LEN + seg + 8 * j;
        float p = exp2f((sc[idx] - m) * 1.4426950408889634f);
        sc[idx] = p;
        sum += p;
    }
    red[row * 8 + seg] = sum;
    __syncthreads();
    if (seg == 0) {
        float s = 0.0f;
        for (int j = 0; j < 8; ++j) s += red[row * 8 + j];
        rinv[row] = 1.0f / s;
    }
    __syncthreads();

    // ---- normalize + coalesced non-temporal stream of attn tile ----
    float* aout = attn_out + ((size_t)(b * HEADS + h) * S_LEN + q0) * S_LEN;
    for (int i = 0; i < 256; ++i) {
        int idx = tid + i * 256;                 // 32*2048 elems, lanes->cols
        float p = sc[idx] * rinv[idx >> 11];
        sc[idx] = p;
        __builtin_nontemporal_store(p, &aout[idx]);
    }
    __syncthreads();

    // ---- heads = P @ V : one 16x16 tile per wave (2 x 4 tiles) ----
    {
        const int mt = wave >> 2, nt = wave & 3;
        v8f acc = zero8();
        const int d = nt * 16 + mrow;            // lane = d_k column
        const bf16_t* vrow = vt_bf + (size_t)(b * DK + d) * S_LEN;
        const float*  arow = sc + (mt * 16 + mrow) * S_LEN;
        for (int k0 = 0; k0 < S_LEN; k0 += 32) {
            v16bf a  = loadA_f32(arow + k0 + kb);
            v16bf bv = cat8(*(const v8bf*)(vrow + k0 + kb),
                            *(const v8bf*)(vrow + k0 + kb + 16));
            acc = wmma_bf16(a, bv, acc);
        }
        int rbase = mt * 16 + (lane >> 4) * 8;
        int gcol  = h * DK + nt * 16 + mrow;     // fused [B,S,H,dk] layout
#pragma unroll
        for (int r = 0; r < 8; ++r)
            heads_bf[(size_t)(b * S_LEN + q0 + rbase + r) * DMODEL + gcol] =
                (bf16_t)acc[r];
    }
}

// ---------------------------------------------------------------------------
extern "C" void kernel_launch(void* const* d_in, const int* in_sizes, int n_in,
                              void* d_out, int out_size, void* d_ws, size_t ws_size,
                              hipStream_t stream)
{
    (void)in_sizes; (void)n_in; (void)out_size; (void)ws_size;

    const float* query = (const float*)d_in[0];
    const float* key   = (const float*)d_in[1];
    const float* value = (const float*)d_in[2];
    const float* w_q   = (const float*)d_in[3];
    const float* b_q   = (const float*)d_in[4];
    const float* w_k   = (const float*)d_in[5];
    const float* b_k   = (const float*)d_in[6];
    const float* w_v   = (const float*)d_in[7];
    const float* b_v   = (const float*)d_in[8];
    const float* w_o   = (const float*)d_in[9];
    const float* b_o   = (const float*)d_in[10];

    float* out  = (float*)d_out;                            // 4096*1024
    float* attn = out + (size_t)MROWS * DMODEL;             // 2*16*2048*2048

    // scratch layout (~17.5 MB total)
    char* ws = (char*)d_ws;
    bf16_t* q_bf  = (bf16_t*)(ws);                          // 8 MB
    bf16_t* k_bf  = (bf16_t*)(ws + 8388608);                // 512 KB
    bf16_t* v_bf  = (bf16_t*)(ws + 8912896);                // 512 KB
    bf16_t* vt_bf = (bf16_t*)(ws + 9437184);                // 512 KB
    bf16_t* heads = (bf16_t*)(ws + 9961472);                // 8 MB

    // Q = query @ w_q + b_q   -> bf16   (4096 x 1024 x 1024)
    gemm_bias_wmma<128, 128, 2, 4, 4, 2, false, true>
        <<<dim3(DMODEL / 128, MROWS / 128), 256, 0, stream>>>(
            query, w_q, b_q, q_bf, MROWS, DMODEL, DMODEL);
    // K = key @ w_k + b_k     -> bf16   (4096 x 64 x 1024)
    gemm_bias_wmma<128, 64, 4, 2, 2, 2, false, true>
        <<<dim3(1, MROWS / 128), 256, 0, stream>>>(
            key, w_k, b_k, k_bf, MROWS, DK, DMODEL);
    // V = value @ w_v + b_v   -> bf16
    gemm_bias_wmma<128, 64, 4, 2, 2, 2, false, true>
        <<<dim3(1, MROWS / 128), 256, 0, stream>>>(
            value, w_v, b_v, v_bf, MROWS, DK, DMODEL);
    // Vt for contiguous B-fragments in P@V
    transpose_v_kernel<<<(BTCH * S_LEN * DK) / 256, 256, 0, stream>>>(v_bf, vt_bf);

    // attention (writes attn + heads)
    size_t smem = 32 * S_LEN * 4 + 32 * DK * 2 + 256 * 4 + 32 * 4 + 32 * 4;
    mqa_attn_kernel<<<dim3(S_LEN / 32, HEADS, BTCH), 256, smem, stream>>>(
        q_bf, k_bf, vt_bf, attn, heads);

    // out = heads @ w_o + b_o -> fp32
    gemm_bias_wmma<128, 128, 2, 4, 4, 2, true, false>
        <<<dim3(DMODEL / 128, MROWS / 128), 256, 0, stream>>>(
            heads, w_o, b_o, out, MROWS, DMODEL, DMODEL);
}
